// Transducer_15625091023607
// MI455X (gfx1250) — matrix operand order repaired and checked by reading, not compiled, not run on hardware
//
#include <hip/hip_runtime.h>

// Problem constants (from reference)
#define B_  8
#define T_  250
#define U_  64
#define V_  1024
#define F_  80
#define H_  512
#define UP1 (U_ + 1)                     // 65
#define M_TOTAL (B_ * T_ * UP1)          // 130000 joint rows
#define MTILES  ((M_TOTAL + 31) / 32)    // 4063 workgroups
#define AROW 520                         // A-panel row stride in ushorts (1040B: 16B aligned, conflict-free)

typedef __attribute__((ext_vector_type(16))) __bf16 v16bf;
typedef __attribute__((ext_vector_type(8)))  float  v8f;

union FragAB { v16bf v; uint4 q[2]; };   // 32 bytes = 8 VGPRs

__device__ __forceinline__ unsigned short f2bf(float f) {
  union { float f; unsigned u; } x; x.f = f;
  unsigned r = x.u + 0x7FFFu + ((x.u >> 16) & 1u);   // round-to-nearest-even
  return (unsigned short)(r >> 16);
}

__device__ __forceinline__ float fast_tanh(float x) {
  x = fminf(15.f, fmaxf(-15.f, x));
  float e = __expf(2.f * x);
  return (e - 1.f) / (e + 1.f);
}

// ---------------------------------------------------------------------------
// 1) encoder GEMM: enc[b*T+t, h] = sum_f inputs[b,t,f] * W_enc[f,h]   (fp32)
// ---------------------------------------------------------------------------
__global__ void enc_gemm(const float* __restrict__ inp,
                         const float* __restrict__ Wenc,
                         float* __restrict__ enc) {
  __shared__ float sIn[F_];
  const int row = blockIdx.x;                 // (b*T + t), 2000 rows
  const int tid = threadIdx.x;
  if (tid < F_) sIn[tid] = inp[(size_t)row * F_ + tid];
  __syncthreads();
  for (int h = tid; h < H_; h += 256) {
    float a = 0.f;
#pragma unroll 8
    for (int f = 0; f < F_; ++f) a += sIn[f] * Wenc[f * H_ + h];
    enc[(size_t)row * H_ + h] = a;
  }
}

// ---------------------------------------------------------------------------
// 2) decoder gather: dec[b,u,:] = emb[padded_target, :]   (fp32)
// ---------------------------------------------------------------------------
__global__ void dec_gather(const int* __restrict__ targets,
                           const float* __restrict__ emb,
                           float* __restrict__ dec) {
  const int bid = blockIdx.x;                 // b*(U+1)+u, 520 rows
  const int b = bid / UP1, u = bid % UP1;
  int tok = (u == 0) ? 0 : targets[b * U_ + (u - 1)];
  tok = (tok < 0) ? 0 : (tok >= V_ ? V_ - 1 : tok);
  for (int h = threadIdx.x; h < H_; h += 256)
    dec[(size_t)bid * H_ + h] = emb[(size_t)tok * H_ + h];
}

// ---------------------------------------------------------------------------
// 3) repack W_joint [H,V] fp32 -> Wt [V,H] bf16 (K contiguous per column,
//    so each lane's B fragment is two contiguous 16B global loads)
// ---------------------------------------------------------------------------
__global__ void repack_wjoint(const float* __restrict__ Wj,
                              unsigned short* __restrict__ Wt) {
  const int idx = blockIdx.x * 256 + threadIdx.x;   // over V_*H_
  const int v = idx >> 9;                           // /H_
  const int k = idx & (H_ - 1);
  Wt[(size_t)v * H_ + k] = f2bf(Wj[(size_t)k * V_ + v]);
}

// ---------------------------------------------------------------------------
// 4) fused joint: per WG 32 rows x 1024 cols.
//    Full A panel (tanh(enc+dec), 32x512 bf16) built once in LDS (one barrier),
//    then a barrier-free K loop of V_WMMA_F32_16X16X32_BF16 with paired,
//    software-pipelined B-fragment loads; in-register log-softmax; emit
//    blank/label log-probs only.
// ---------------------------------------------------------------------------
__global__ void __launch_bounds__(256)
joint_softmax(const float* __restrict__ enc, const float* __restrict__ dec,
              const unsigned short* __restrict__ Wt,
              const float* __restrict__ bj,
              const int* __restrict__ targets,
              float* __restrict__ blank_lp, float* __restrict__ lab_lp) {
  __shared__ __align__(16) unsigned short sAp[32 * AROW];   // 33,280 B
  __shared__ int rB[32], rT[32], rU[32], rLab[32];
  __shared__ float sRed[32 * 8];
  __shared__ float sRowMax[32], sLse[32];

  const int tid  = threadIdx.x;
  const int lane = tid & 31;
  const int wv   = tid >> 5;                    // wave id 0..7 -> cols wv*128..+127
  const int m0   = blockIdx.x * 32;

  if (tid < 32) {
    const int m = m0 + tid;
    const bool valid = m < M_TOTAL;
    const int mm = valid ? m : 0;
    const int b = mm / (T_ * UP1);
    const int rem = mm - b * (T_ * UP1);
    const int t = rem / UP1;
    const int u = rem - t * UP1;
    rB[tid] = valid ? b : -1;
    rT[tid] = t;  rU[tid] = u;
    int lab = -1;
    if (valid && u < U_) {
      lab = targets[b * U_ + u];
      lab = (lab < 0) ? 0 : (lab >= V_ ? V_ - 1 : lab);
    }
    rLab[tid] = lab;
  }
  __syncthreads();

  // ---- build full A panel: each thread fills (row ar, K = ak + 32*j) ----
  const int ar = tid >> 3;                  // 0..31
  const int ak = (tid & 7) * 4;             // 0..28
  const bool rowValid = (rB[ar] >= 0);
  const size_t encOff = (size_t)(rowValid ? (rB[ar] * T_ + rT[ar]) : 0) * H_ + ak;
  const size_t decOff = (size_t)(rowValid ? (rB[ar] * UP1 + rU[ar]) : 0) * H_ + ak;

#pragma unroll 4
  for (int j = 0; j < 16; ++j) {
    const int kb = j * 32;
    float4 ev = make_float4(0.f, 0.f, 0.f, 0.f);
    float4 dv = make_float4(0.f, 0.f, 0.f, 0.f);
    if (rowValid) {
      ev = *(const float4*)(enc + encOff + kb);
      dv = *(const float4*)(dec + decOff + kb);
    }
    unsigned short* p = &sAp[ar * AROW + ak + kb];
    p[0] = f2bf(fast_tanh(ev.x + dv.x));
    p[1] = f2bf(fast_tanh(ev.y + dv.y));
    p[2] = f2bf(fast_tanh(ev.z + dv.z));
    p[3] = f2bf(fast_tanh(ev.w + dv.w));
  }
  __syncthreads();                          // single barrier before GEMM

  v8f acc[2][8] = {};                       // 16 C/D frags (fp32 accumulate)

  // Per-lane fragment bases (ISA 7.12.2 layouts):
  //  A 16x32 bf16: lane=row (l&15), half h=l>>4: e<8 -> K=h*8+e ; e>=8 -> K=16+h*8+(e-8)
  //  B 32x16 bf16: lane=col (l&15), half h=l>>4: e<8 -> K=h*8+e ; e>=8 -> K=16+h*8+(e-8)
  const unsigned short* WtBase =
      Wt + (size_t)(wv * 128 + (lane & 15)) * H_ + ((lane >> 4) & 1) * 8;
  const unsigned short* sApBase = sAp + (lane & 15) * AROW + ((lane >> 4) & 1) * 8;

#pragma unroll 1
  for (int ks = 0; ks < 16; ++ks) {
    const int kb = ks * 32;
    FragAB a0, a1;
    a0.q[0] = *(const uint4*)(sApBase + kb);
    a0.q[1] = *(const uint4*)(sApBase + kb + 16);
    a1.q[0] = *(const uint4*)(sApBase + 16 * AROW + kb);
    a1.q[1] = *(const uint4*)(sApBase + 16 * AROW + kb + 16);

    const unsigned short* wp = WtBase + kb;
#pragma unroll
    for (int nt = 0; nt < 8; nt += 2) {
      const unsigned short* pb0 = wp + (size_t)nt * 16 * H_;
      const unsigned short* pb1 = wp + (size_t)(nt + 1) * 16 * H_;
      __builtin_prefetch(pb0 + 32, 0, 1);   // next K-slab, same column strips
      __builtin_prefetch(pb1 + 32, 0, 1);
      FragAB b0, b1;                        // distinct regs -> 4 loads in flight
      b0.q[0] = *(const uint4*)(pb0);
      b0.q[1] = *(const uint4*)(pb0 + 16);
      b1.q[0] = *(const uint4*)(pb1);
      b1.q[1] = *(const uint4*)(pb1 + 16);
      acc[0][nt] = __builtin_amdgcn_wmma_f32_16x16x32_bf16(
          false, a0.v, false, b0.v, (short)0, acc[0][nt], false, false);
      acc[1][nt] = __builtin_amdgcn_wmma_f32_16x16x32_bf16(
          false, a1.v, false, b0.v, (short)0, acc[1][nt], false, false);
      acc[0][nt + 1] = __builtin_amdgcn_wmma_f32_16x16x32_bf16(
          false, a0.v, false, b1.v, (short)0, acc[0][nt + 1], false, false);
      acc[1][nt + 1] = __builtin_amdgcn_wmma_f32_16x16x32_bf16(
          false, a1.v, false, b1.v, (short)0, acc[1][nt + 1], false, false);
    }
  }

  // bias add (col = wv*128 + nt*16 + (lane&15) for every accum element)
  float bias[8];
#pragma unroll
  for (int nt = 0; nt < 8; ++nt) bias[nt] = bj[wv * 128 + nt * 16 + (lane & 15)];
#pragma unroll
  for (int mt = 0; mt < 2; ++mt)
#pragma unroll
    for (int nt = 0; nt < 8; ++nt)
#pragma unroll
      for (int i = 0; i < 8; ++i) acc[mt][nt][i] += bias[nt];

  // ---- row max (C/D layout: VGPR i -> row mt*16+i (+8 for upper lanes)) ----
  __syncthreads();
#pragma unroll
  for (int mt = 0; mt < 2; ++mt)
#pragma unroll
    for (int i = 0; i < 8; ++i) {
      float v = acc[mt][0][i];
#pragma unroll
      for (int nt = 1; nt < 8; ++nt) v = fmaxf(v, acc[mt][nt][i]);
      v = fmaxf(v, __shfl_xor(v, 1, 32));
      v = fmaxf(v, __shfl_xor(v, 2, 32));
      v = fmaxf(v, __shfl_xor(v, 4, 32));
      v = fmaxf(v, __shfl_xor(v, 8, 32));
      const int row = mt * 16 + i + ((lane & 16) ? 8 : 0);
      if ((lane & 15) == 0) sRed[row * 8 + wv] = v;
    }
  __syncthreads();
  if (tid < 32) {
    float v = sRed[tid * 8];
    for (int w = 1; w < 8; ++w) v = fmaxf(v, sRed[tid * 8 + w]);
    sRowMax[tid] = v;
  }
  __syncthreads();

  // ---- row sum of exp ----
#pragma unroll
  for (int mt = 0; mt < 2; ++mt)
#pragma unroll
    for (int i = 0; i < 8; ++i) {
      const int row = mt * 16 + i + ((lane & 16) ? 8 : 0);
      const float rm = sRowMax[row];
      float s = 0.f;
#pragma unroll
      for (int nt = 0; nt < 8; ++nt) s += __expf(acc[mt][nt][i] - rm);
      s += __shfl_xor(s, 1, 32);
      s += __shfl_xor(s, 2, 32);
      s += __shfl_xor(s, 4, 32);
      s += __shfl_xor(s, 8, 32);
      if ((lane & 15) == 0) sRed[row * 8 + wv] = s;
    }
  __syncthreads();
  if (tid < 32) {
    float s = 0.f;
    for (int w = 0; w < 8; ++w) s += sRed[tid * 8 + w];
    sLse[tid] = sRowMax[tid] + __logf(s);
  }
  __syncthreads();

  // ---- emit blank (col 0) and label (col = target) log-probs ----
#pragma unroll
  for (int mt = 0; mt < 2; ++mt)
#pragma unroll
    for (int i = 0; i < 8; ++i) {
      const int row = mt * 16 + i + ((lane & 16) ? 8 : 0);
      if (rB[row] < 0) continue;
      const float lse = sLse[row];
      const int m = m0 + row;                      // == (b*T+t)*(U+1)+u
#pragma unroll
      for (int nt = 0; nt < 8; ++nt) {
        const int col = wv * 128 + nt * 16 + (lane & 15);
        const float lp = acc[mt][nt][i] - lse;
        if (col == 0) blank_lp[m] = lp;
        if (col == rLab[row]) {
          const int b = rB[row], t = rT[row], u = rU[row];
          lab_lp[((size_t)(b * T_ + t)) * U_ + u] = lp;
        }
      }
    }
}

// ---------------------------------------------------------------------------
// 5) RNNT forward DP per batch: anti-diagonal wavefront in LDS ping-pong
// ---------------------------------------------------------------------------
#define NEGINF (-1e30f)
__global__ void rnnt_dp(const float* __restrict__ blank_lp,
                        const float* __restrict__ lab_lp,
                        const int* __restrict__ in_len,
                        const int* __restrict__ tg_len,
                        float* __restrict__ losses) {
  __shared__ float dbuf[2][UP1];
  __shared__ float res;
  const int b = blockIdx.x;
  const int tid = threadIdx.x;
  const int Tb = in_len[b];
  const int Ub = tg_len[b];
  const float* BL = blank_lp + (size_t)b * T_ * UP1;
  const float* LA = lab_lp + (size_t)b * T_ * U_;
  if (tid == 0) { dbuf[0][0] = 0.f; res = NEGINF; }
  for (int d = 1; d <= (T_ - 1) + U_; ++d) {
    __syncthreads();
    const int cur = d & 1, prev = cur ^ 1;
    const int ulo = (d - (T_ - 1) > 0) ? d - (T_ - 1) : 0;
    const int uhi = (d < U_) ? d : U_;
    for (int u = ulo + tid; u <= uhi; u += blockDim.x) {
      const int t = d - u;
      const float a1 = (t > 0) ? dbuf[prev][u]     + BL[(t - 1) * UP1 + u]   : NEGINF;
      const float a2 = (u > 0) ? dbuf[prev][u - 1] + LA[t * U_ + (u - 1)]    : NEGINF;
      const float mx = fmaxf(a1, a2);
      const float val = (mx <= 0.5f * NEGINF)
                            ? NEGINF
                            : mx + __logf(__expf(a1 - mx) + __expf(a2 - mx));
      dbuf[cur][u] = val;
      if (t == Tb - 1 && u == Ub) res = val + BL[(Tb - 1) * UP1 + u];
    }
  }
  __syncthreads();
  if (tid == 0) losses[b] = -res;
}

// ---------------------------------------------------------------------------
// 6) mean over batch
// ---------------------------------------------------------------------------
__global__ void finalize(const float* __restrict__ losses, float* __restrict__ out) {
  if (threadIdx.x == 0) {
    float s = 0.f;
    for (int b = 0; b < B_; ++b) s += losses[b];
    out[0] = s / (float)B_;
  }
}

// ---------------------------------------------------------------------------
extern "C" void kernel_launch(void* const* d_in, const int* in_sizes, int n_in,
                              void* d_out, int out_size, void* d_ws, size_t ws_size,
                              hipStream_t stream) {
  (void)in_sizes; (void)n_in; (void)out_size; (void)ws_size;
  const float* inputs  = (const float*)d_in[0];
  const int*   targets = (const int*)d_in[1];
  const int*   in_len  = (const int*)d_in[2];
  const int*   tg_len  = (const int*)d_in[3];
  const float* W_enc   = (const float*)d_in[4];
  const float* emb     = (const float*)d_in[5];
  const float* W_joint = (const float*)d_in[6];
  const float* b_joint = (const float*)d_in[7];
  float* out = (float*)d_out;

  char* ws = (char*)d_ws;
  size_t off = 0;
  float* enc = (float*)(ws + off);            off += (size_t)B_ * T_ * H_ * 4;   // 4.10 MB
  float* dec = (float*)(ws + off);            off += (size_t)B_ * UP1 * H_ * 4;  // 1.06 MB
  unsigned short* Wt = (unsigned short*)(ws + off); off += (size_t)V_ * H_ * 2;  // 1.05 MB
  float* blank_lp = (float*)(ws + off);       off += (((size_t)M_TOTAL * 4) + 255) & ~(size_t)255;
  float* lab_lp = (float*)(ws + off);         off += (size_t)B_ * T_ * U_ * 4;
  float* losses = (float*)(ws + off);         off += 256;

  enc_gemm<<<B_ * T_, 256, 0, stream>>>(inputs, W_enc, enc);
  dec_gather<<<B_ * UP1, 256, 0, stream>>>(targets, emb, dec);
  repack_wjoint<<<(V_ * H_) / 256, 256, 0, stream>>>(W_joint, Wt);
  joint_softmax<<<MTILES, 256, 0, stream>>>(enc, dec, Wt, b_joint, targets,
                                            blank_lp, lab_lp);
  rnnt_dp<<<B_, 128, 0, stream>>>(blank_lp, lab_lp, in_len, tg_len, losses);
  finalize<<<1, 64, 0, stream>>>(losses, out);
}
